// SequenceSplitter_27281632264458
// MI455X (gfx1250) — matrix operand order repaired and checked by reading, not compiled, not run on hardware
//
#include <hip/hip_runtime.h>
#include <hip/hip_bf16.h>

#define NN   2048
#define SEQ  128
#define E    512
#define R    128

typedef __attribute__((ext_vector_type(16))) __bf16    v16bf;
typedef __attribute__((ext_vector_type(8)))  float     v8f;
typedef __attribute__((ext_vector_type(4)))  unsigned  v4u;

union ABFrag { v16bf v; unsigned u[8]; v4u q[2]; };

__device__ __forceinline__ unsigned f2bf_bits(float x) {
  unsigned u = __float_as_uint(x);
  return (u + 0x7fffu + ((u >> 16) & 1u)) >> 16;   // round-to-nearest-even bf16
}
__device__ __forceinline__ unsigned pack2(float lo, float hi) {
  return f2bf_bits(lo) | (f2bf_bits(hi) << 16);
}

// Branch-free tanh: 1 - 2/(e^{2x}+1); saturates to +/-1, no exec divergence.
__device__ __forceinline__ float tanh_fast(float x) {
  float e = __expf(2.f * x);
  return 1.f - 2.f * __builtin_amdgcn_rcpf(e + 1.f);
}

// A fragment, 16x32 bf16: lane l holds row M=l%16;
// lanes 0-15: VGPR0-7 K={0..15}; lanes16-31: K={16..31}.
// Packed pairs are K-contiguous -> two 16B LDS loads.
__device__ __forceinline__ v16bf load_a_pk(const unsigned* buf, int strideU,
                                           int row, int kbase, int half) {
  ABFrag f;
  const unsigned* p = buf + row * strideU + ((kbase + 16 * half) >> 1);
  f.q[0] = *(const v4u*)p;
  f.q[1] = *(const v4u*)(p + 4);
  return f.v;
}

// B fragment from pre-swizzled bf16 weights: layout [kc][col][halfk][v] of
// packed pairs -> each lane's 8 VGPRs are 32 contiguous bytes.
__device__ __forceinline__ v16bf load_b_sw(const unsigned* __restrict__ Ws,
                                           int kc, int col, int halfk) {
  ABFrag f;
  const unsigned* p = Ws + (((kc * 128 + col) * 2 + halfk) << 3);
  f.q[0] = *(const v4u*)p;
  f.q[1] = *(const v4u*)(p + 4);
  return f.v;
}

// B fragment from bf16 row-major Y (as packed u32, row stride R/2 u32):
// element (k, n) = Y[row n][k] -> contiguous K, two 16B loads.
__device__ __forceinline__ v16bf load_b_bf(const unsigned* __restrict__ Yu,
                                           int row, int kbase, int halfk) {
  ABFrag f;
  const unsigned* p = Yu + row * (R / 2) + ((kbase + 16 * halfk) >> 1);
  f.q[0] = *(const v4u*)p;
  f.q[1] = *(const v4u*)(p + 4);
  return f.v;
}

__device__ __forceinline__ v8f wmma_bf16(v16bf a, v16bf b, v8f c) {
  return __builtin_amdgcn_wmma_f32_16x16x32_bf16(false, a, false, b,
                                                 (short)0, c, false, false);
}

// kc-outer / nt-inner GEMM: one A fragment feeds 8 back-to-back WMMAs.
__device__ __forceinline__ void gemm_acc(const unsigned* Abuf, int strideU,
                                         int arow, int half, int colB,
                                         const unsigned* __restrict__ Ws,
                                         int kchunks, v8f c[8]) {
  for (int kc = 0; kc < kchunks; ++kc) {
    v16bf a = load_a_pk(Abuf, strideU, arow, kc * 32, half);
    #pragma unroll
    for (int nt = 0; nt < 8; ++nt) {
      v16bf b = load_b_sw(Ws, kc, nt * 16 + colB, half);
      c[nt] = wmma_bf16(a, b, c[nt]);
    }
  }
}

__global__ void zero_kernel(float* p, int count) {
  int i = blockIdx.x * blockDim.x + threadIdx.x;
  if (i < count) p[i] = 0.f;
}

// Convert fp32 weight (kdim x R) into bf16 B-fragment-swizzled layout.
__global__ void swizzle_w_kernel(const float* __restrict__ W,
                                 unsigned* __restrict__ Ws, int kdim) {
  int i = blockIdx.x * blockDim.x + threadIdx.x;   // output u32 index
  if (i >= kdim * R / 2) return;
  int v = i & 7, h = (i >> 3) & 1, col = (i >> 4) & 127, kc = i >> 11;
  int k = kc * 32 + 16 * h + 2 * v;
  Ws[i] = pack2(W[k * R + col], W[(k + 1) * R + col]);
}

// One block per sequence n. 256 threads = 8 waves; wave w owns M rows [16w,16w+16).
__global__ void __launch_bounds__(256)
fused_mlp_kernel(const int* __restrict__ C, const float* __restrict__ L,
                 const float* __restrict__ PW, const float* __restrict__ NW,
                 const unsigned* __restrict__ W1ps, const float* __restrict__ b1p,
                 const unsigned* __restrict__ W2ps, const float* __restrict__ b2p,
                 const unsigned* __restrict__ W1ns, const float* __restrict__ b1n,
                 const unsigned* __restrict__ W2ns, const float* __restrict__ b2n,
                 const unsigned* __restrict__ Was,  const float* __restrict__ ba,
                 float* __restrict__ accp, float* __restrict__ accn,
                 float* __restrict__ accf) {
  __shared__ unsigned       Lbf[(SEQ * E) / 2];   // 131072 B: L[n] in bf16
  __shared__ unsigned short Hbs[SEQ * R];         //  32768 B: hidden layer bf16

  const int n    = blockIdx.x;
  const int tid  = threadIdx.x;
  const int wave = tid >> 5, lane = tid & 31;
  const int mbase = wave * 16;
  const int half  = lane >> 4;      // K-half for A/B, also D row-half
  const int colB  = lane & 15;      // B/D column within tile
  const int arow  = mbase + (lane & 15);

  // Stage L[n] fp32 -> bf16 in LDS (coalesced b128 loads).
  const float* Ln = L + (size_t)n * (SEQ * E);
  for (int i = tid; i < (SEQ * E) / 4; i += 256) {
    float4 f = ((const float4*)Ln)[i];
    Lbf[2 * i]     = pack2(f.x, f.y);
    Lbf[2 * i + 1] = pack2(f.z, f.w);
  }

  // Row weights for the 8 D-rows this lane owns (mask commutes through GEMM).
  const int cs = C[2 * n], ce = C[2 * n + 1];
  float wp[8], wn[8];
  #pragma unroll
  for (int r = 0; r < 8; ++r) {
    int t = mbase + r + 8 * half;
    float pw = PW[n * SEQ + t], nw = NW[n * SEQ + t];
    bool in = (t >= cs) && (t <= ce);
    wp[r] = in ? pw : 0.f;
    wn[r] = in ? 0.f : nw;
  }

  __syncthreads();

  const unsigned* Hu = (const unsigned*)Hbs;

  // ---- pos / neg branches: t = mean_rows(tanh(tanh(diag(w) L W1 + b1) W2 + b2))
  for (int br = 0; br < 2; ++br) {
    const unsigned* W1s = br ? W1ns : W1ps;  const float* b1 = br ? b1n : b1p;
    const unsigned* W2s = br ? W2ns : W2ps;  const float* b2 = br ? b2n : b2p;
    float* acc = br ? accn : accp;

    {  // GEMM1 (128x512 @ 512x128); row-scale applied to accumulator pre-tanh.
      v8f c[8] = {};
      gemm_acc(Lbf, E / 2, arow, half, colB, W1s, E / 32, c);
      #pragma unroll
      for (int nt = 0; nt < 8; ++nt) {
        int col = nt * 16 + colB;
        float bias = b1[col];
        #pragma unroll
        for (int r = 0; r < 8; ++r) {
          float w = br ? wn[r] : wp[r];
          float h = tanh_fast(c[nt][r] * w + bias);
          int m = mbase + r + 8 * half;
          Hbs[m * R + col] = (unsigned short)f2bf_bits(h);
        }
      }
    }
    __syncthreads();

    {  // GEMM2 (128x128 @ 128x128) + tanh + column mean (atomic accumulate).
      v8f c[8] = {};
      gemm_acc(Hu, R / 2, arow, half, colB, W2s, R / 32, c);
      #pragma unroll
      for (int nt = 0; nt < 8; ++nt) {
        int col = nt * 16 + colB;
        float bias = b2[col];
        float s = 0.f;
        #pragma unroll
        for (int r = 0; r < 8; ++r) s += tanh_fast(c[nt][r] + bias);
        s += __shfl_xor(s, 16);          // lane & lane+16 share a column
        if (half == 0) atomicAdd(&acc[n * R + col], s);
      }
    }
    __syncthreads();
  }

  // ---- final_feature: mean_rows(tanh(diag(wp+wn) L Wa + ba))
  {
    v8f c[8] = {};
    gemm_acc(Lbf, E / 2, arow, half, colB, Was, E / 32, c);
    #pragma unroll
    for (int nt = 0; nt < 8; ++nt) {
      int col = nt * 16 + colB;
      float bias = ba[col];
      float s = 0.f;
      #pragma unroll
      for (int r = 0; r < 8; ++r)
        s += tanh_fast(c[nt][r] * (wp[r] + wn[r]) + bias);
      s += __shfl_xor(s, 16);
      if (half == 0) atomicAdd(&accf[n * R + col], s);
    }
  }
}

// Divide sums by SEQ, emit bf16 pos_t/neg_t + fp32 norms + final_feature.
__global__ void __launch_bounds__(128)
finalize_kernel(const float* __restrict__ accp, const float* __restrict__ accn,
                const float* __restrict__ accf,
                unsigned short* __restrict__ posbf, unsigned short* __restrict__ negbf,
                float* __restrict__ posnorm, float* __restrict__ negnorm,
                float* __restrict__ out_feat) {
  int n = blockIdx.x, j = threadIdx.x;
  float p = accp[n * R + j] * (1.f / SEQ);
  float q = accn[n * R + j] * (1.f / SEQ);
  float f = accf[n * R + j] * (1.f / SEQ);
  posbf[n * R + j] = (unsigned short)f2bf_bits(p);
  negbf[n * R + j] = (unsigned short)f2bf_bits(q);
  out_feat[n * R + j] = f;

  float pp = p * p, qq = q * q;
  #pragma unroll
  for (int off = 16; off; off >>= 1) {
    pp += __shfl_xor(pp, off);
    qq += __shfl_xor(qq, off);
  }
  __shared__ float red[2][4];
  int w = j >> 5;
  if ((j & 31) == 0) { red[0][w] = pp; red[1][w] = qq; }
  __syncthreads();
  if (j == 0) {
    posnorm[n] = red[0][0] + red[0][1] + red[0][2] + red[0][3];
    negnorm[n] = red[1][0] + red[1][1] + red[1][2] + red[1][3];
  }
}

// 16x16 Gram tile (K=128) via 4 bf16 WMMAs, then sum(exp(-safe_cdist)).
// blockIdx.y: 0 -> pos@pos, 1 -> pos@neg. 1 wave per block.
__global__ void __launch_bounds__(32)
sim_kernel(const unsigned* __restrict__ Xu, const unsigned* __restrict__ Yup,
           const unsigned* __restrict__ Yun, const float* __restrict__ nx,
           const float* __restrict__ nyp, const float* __restrict__ nyn,
           float* __restrict__ sums) {
  int tile  = blockIdx.x;       // 0..(128*128-1)
  int which = blockIdx.y;
  int it = tile & 127, jt = tile >> 7;
  const unsigned* Yu = which ? Yun : Yup;
  const float*    ny = which ? nyn : nyp;
  int lane = threadIdx.x;
  int half = lane >> 4, colB = lane & 15;

  v8f c = {};
  #pragma unroll
  for (int kc = 0; kc < R / 32; ++kc) {
    v16bf a = load_a_pk(Xu, R / 2, it * 16 + colB, kc * 32, half);
    v16bf b = load_b_bf(Yu, jt * 16 + colB, kc * 32, half);
    c = wmma_bf16(a, b, c);
  }

  float nyj = ny[jt * 16 + colB];
  float s = 0.f;
  #pragma unroll
  for (int r = 0; r < 8; ++r) {
    int i = it * 16 + r + 8 * half;
    float d2 = nx[i] + nyj - 2.f * c[r];
    d2 = fmaxf(d2, 0.f);
    float d = (d2 > 0.f) ? sqrtf(d2) : 0.f;
    s += __expf(-d);
  }
  #pragma unroll
  for (int off = 16; off; off >>= 1) s += __shfl_xor(s, off);
  if (lane == 0) atomicAdd(&sums[which], s);
}

__global__ void loss_kernel(const float* __restrict__ sums, float* __restrict__ out) {
  out[NN * R] = -(logf(sums[0]) - logf(sums[1]));
}

extern "C" void kernel_launch(void* const* d_in, const int* in_sizes, int n_in,
                              void* d_out, int out_size, void* d_ws, size_t ws_size,
                              hipStream_t stream) {
  const int*   C   = (const int*)d_in[0];
  const float* L   = (const float*)d_in[1];
  const float* PW  = (const float*)d_in[2];
  const float* NW  = (const float*)d_in[3];
  const float* W1p = (const float*)d_in[4];
  const float* b1p = (const float*)d_in[5];
  const float* W2p = (const float*)d_in[6];
  const float* b2p = (const float*)d_in[7];
  const float* W1n = (const float*)d_in[8];
  const float* b1n = (const float*)d_in[9];
  const float* W2n = (const float*)d_in[10];
  const float* b2n = (const float*)d_in[11];
  const float* Wa  = (const float*)d_in[12];
  const float* ba  = (const float*)d_in[13];
  float* out = (float*)d_out;

  // Workspace layout (all sections 16B-aligned).
  char* ws = (char*)d_ws;
  float* accp = (float*)ws;                                  // NN*R f32
  float* accn = accp + NN * R;
  float* accf = accn + NN * R;
  unsigned short* posbf = (unsigned short*)(accf + NN * R);  // NN*R bf16
  unsigned short* negbf = posbf + NN * R;
  unsigned* W1ps = (unsigned*)(negbf + NN * R);              // E*R/2 u32
  unsigned* W2ps = W1ps + E * R / 2;                         // R*R/2 u32
  unsigned* W1ns = W2ps + R * R / 2;
  unsigned* W2ns = W1ns + E * R / 2;
  unsigned* Was  = W2ns + R * R / 2;
  float* posnorm = (float*)(Was + E * R / 2);                // NN f32
  float* negnorm = posnorm + NN;
  float* sums    = negnorm + NN;                             // 2 f32

  // 1) init accumulators + sums
  int zc = 3 * NN * R;
  zero_kernel<<<(zc + 255) / 256, 256, 0, stream>>>(accp, zc);
  zero_kernel<<<1, 32, 0, stream>>>(sums, 2);

  // 2) pre-swizzle weights into bf16 B-fragment layout (shared by all blocks)
  swizzle_w_kernel<<<(E * R / 2 + 255) / 256, 256, 0, stream>>>(W1p, W1ps, E);
  swizzle_w_kernel<<<(R * R / 2 + 255) / 256, 256, 0, stream>>>(W2p, W2ps, R);
  swizzle_w_kernel<<<(E * R / 2 + 255) / 256, 256, 0, stream>>>(W1n, W1ns, E);
  swizzle_w_kernel<<<(R * R / 2 + 255) / 256, 256, 0, stream>>>(W2n, W2ns, R);
  swizzle_w_kernel<<<(E * R / 2 + 255) / 256, 256, 0, stream>>>(Wa, Was, E);

  // 3) fused masked MLPs + final feature accumulation
  fused_mlp_kernel<<<NN, 256, 0, stream>>>(C, L, PW, NW, W1ps, b1p, W2ps, b2p,
                                           W1ns, b1n, W2ns, b2n, Was, ba,
                                           accp, accn, accf);

  // 4) means, bf16 embeddings, norms, final_feature output
  finalize_kernel<<<NN, 128, 0, stream>>>(accp, accn, accf, posbf, negbf,
                                          posnorm, negnorm, out);

  // 5) pairwise similarity sums (pos-pos and pos-neg)
  sim_kernel<<<dim3((NN / 16) * (NN / 16), 2), 32, 0, stream>>>(
      (const unsigned*)posbf, (const unsigned*)posbf, (const unsigned*)negbf,
      posnorm, posnorm, negnorm, sums);

  // 6) loss scalar
  loss_kernel<<<1, 1, 0, stream>>>(sums, out);
}
